// TaskLinear_85950885527990
// MI455X (gfx1250) — matrix operand order repaired and checked by reading, not compiled, not run on hardware
//
#include <hip/hip_runtime.h>
#include <hip/hip_bf16.h>

#define NUM_TASKS 64
#define IN_F      512
#define OUT_F     512
#define BATCH     2048

typedef __attribute__((ext_vector_type(16))) __bf16 v16bf;
typedef __attribute__((ext_vector_type(8)))  float  v8f;

// ---------------- bucket-by-task kernels ----------------

__global__ void tl_zero_counts(int* counts) {
    if (threadIdx.x < NUM_TASKS) counts[threadIdx.x] = 0;
}

__global__ void tl_count(const int* __restrict__ task_id,
                         int* __restrict__ counts,
                         int* __restrict__ slot) {
    int i = blockIdx.x * blockDim.x + threadIdx.x;
    if (i < BATCH) {
        int t = task_id[i];
        slot[i] = atomicAdd(&counts[t], 1);
    }
}

__global__ void tl_scan(const int* __restrict__ counts, int* __restrict__ offsets) {
    if (threadIdx.x == 0 && blockIdx.x == 0) {
        int acc = 0;
        for (int t = 0; t < NUM_TASKS; ++t) {
            offsets[t] = acc;
            acc += counts[t];
        }
    }
}

__global__ void tl_scatter(const int* __restrict__ task_id,
                           const int* __restrict__ offsets,
                           const int* __restrict__ slot,
                           int* __restrict__ sorted) {
    int i = blockIdx.x * blockDim.x + threadIdx.x;
    if (i < BATCH) {
        int t = task_id[i];
        sorted[offsets[t] + slot[i]] = i;
    }
}

// ---------------- grouped GEMM kernel ----------------
// Grid: (NUM_TASKS, OUT_F/128). Block: 256 threads = 8 waves.
// Wave w computes output columns [blockIdx.y*128 + w*16, +16) for all samples
// of its task, 16 samples per iteration, via v_wmma_f32_16x16x32_bf16.

__global__ __launch_bounds__(256)
void tl_gemm(const float* __restrict__ x,
             const float* __restrict__ w,
             const float* __restrict__ bias,
             const int*   __restrict__ counts,
             const int*   __restrict__ offsets,
             const int*   __restrict__ sorted,
             float* __restrict__ out) {
    const int task = blockIdx.x;
    const int cnt  = counts[task];
    if (cnt == 0) return;                 // block-uniform
    const int off  = offsets[task];

    const int lane = threadIdx.x & 31;
    const int wave = threadIdx.x >> 5;
    const int half = lane >> 4;           // 0: lanes 0-15, 1: lanes 16-31
    const int l16  = lane & 15;

    const int obase = blockIdx.y * 128 + wave * 16;
    const int ocol  = obase + l16;        // this lane's output feature (B column / D column)
    const float* wrow = w + ((size_t)task * OUT_F + ocol) * IN_F;
    const float  bv   = bias[(size_t)task * OUT_F + ocol];

    // A tile (16 samples x 512) staged in LDS as bf16, shared by all 8 waves.
    __shared__ __bf16 As[16][IN_F + 8];

    for (int s0 = 0; s0 < cnt; s0 += 16) {
        __syncthreads();  // previous tile fully consumed

        // Cooperative fill: thread tid loads row tid/16, 32 cols at (tid%16)*32.
        {
            const int r  = threadIdx.x >> 4;
            const int c0 = (threadIdx.x & 15) * 32;
            const int sr = s0 + r;
            const int xrow = sorted[off + (sr < cnt ? sr : cnt - 1)];
            const float4* xp = (const float4*)(x + (size_t)xrow * IN_F + c0);
            #pragma unroll
            for (int j = 0; j < 8; ++j) {
                float4 v = xp[j];
                As[r][c0 + 4*j + 0] = (__bf16)v.x;
                As[r][c0 + 4*j + 1] = (__bf16)v.y;
                As[r][c0 + 4*j + 2] = (__bf16)v.z;
                As[r][c0 + 4*j + 3] = (__bf16)v.w;
            }
        }
        __syncthreads();

        v8f acc = {};
        #pragma unroll 4
        for (int k = 0; k < IN_F; k += 32) {
            // A fragment (16x32 bf16): lane holds row M=l16;
            // elems 0..7  = K in [k + half*8,      +8)
            // elems 8..15 = K in [k + 16 + half*8, +8)
            v16bf a;
            {
                const __bf16* ap0 = &As[l16][k + half * 8];
                const __bf16* ap1 = &As[l16][k + 16 + half * 8];
                #pragma unroll
                for (int j = 0; j < 8; ++j) a[j]     = ap0[j];
                #pragma unroll
                for (int j = 0; j < 8; ++j) a[8 + j] = ap1[j];
            }
            // B fragment (32x16 bf16): lane holds column N=l16;
            // elems 0..7 = K in [k + half*8, +8), elems 8..15 = K in [k + 16 + half*8, +8)  (W row `ocol` is B column N)
            v16bf b;
            {
                const float* bp0 = wrow + k + half * 8;
                const float* bp1 = wrow + k + 16 + half * 8;
                #pragma unroll
                for (int j = 0; j < 8; ++j) b[j] = (__bf16)bp0[j];
                #pragma unroll
                for (int j = 0; j < 8; ++j) b[8 + j] = (__bf16)bp1[j];
            }
            acc = __builtin_amdgcn_wmma_f32_16x16x32_bf16(
                /*neg_a=*/false, a, /*neg_b=*/false, b,
                /*c_mod=*/(short)0, acc, /*reuse_a=*/false, /*reuse_b=*/false);
        }

        // D layout: lane gives N (=ocol), VGPR r gives M = r + 8*half.
        #pragma unroll
        for (int r = 0; r < 8; ++r) {
            const int m = r + half * 8;
            const int s = s0 + m;
            if (s < cnt) {
                const int yrow = sorted[off + s];
                out[(size_t)yrow * OUT_F + ocol] = acc[r] + bv;
            }
        }
    }
}

// ---------------- host-side launch ----------------

extern "C" void kernel_launch(void* const* d_in, const int* in_sizes, int n_in,
                              void* d_out, int out_size, void* d_ws, size_t ws_size,
                              hipStream_t stream) {
    const float* x       = (const float*)d_in[0];   // (2048, 512) f32
    const int*   task_id = (const int*)  d_in[1];   // (2048,) i32
    const float* weights = (const float*)d_in[2];   // (64, 512, 512) f32
    const float* bias    = (const float*)d_in[3];   // (64, 512) f32
    float* out = (float*)d_out;                     // (2048, 512) f32

    int* counts  = (int*)d_ws;          // 64
    int* offsets = counts + NUM_TASKS;  // 64
    int* slot    = offsets + NUM_TASKS; // 2048
    int* sorted  = slot + BATCH;        // 2048

    tl_zero_counts<<<1, 64, 0, stream>>>(counts);
    tl_count<<<(BATCH + 255) / 256, 256, 0, stream>>>(task_id, counts, slot);
    tl_scan<<<1, 1, 0, stream>>>(counts, offsets);
    tl_scatter<<<(BATCH + 255) / 256, 256, 0, stream>>>(task_id, offsets, slot, sorted);

    dim3 grid(NUM_TASKS, OUT_F / 128);
    tl_gemm<<<grid, 256, 0, stream>>>(x, weights, bias, counts, offsets, sorted, out);
}
